// SpatialConv_3607772529205
// MI455X (gfx1250) — matrix operand order repaired and checked by reading, not compiled, not run on hardware
//
#include <hip/hip_runtime.h>
#include <hip/hip_bf16.h>
#include <stdint.h>

#define NNODE 50000
#define NEDGE 800000
#define HDIM  128
#define EDDIM 64

typedef __attribute__((ext_vector_type(16))) __bf16 v16bf;
typedef __attribute__((ext_vector_type(8)))  float  v8f;

union BF16Frag { v16bf v; uint4 q[2]; };

__device__ __forceinline__ unsigned short f2bf(float f) {
    unsigned u = __float_as_uint(f);
    unsigned r = (u + 0x7FFFu + ((u >> 16) & 1u)) >> 16;   // round-to-nearest-even
    return (unsigned short)r;
}
__device__ __forceinline__ float bf2f(unsigned short b) {
    return __uint_as_float(((unsigned)b) << 16);
}

// One 16x16x32 bf16 WMMA step. Arow: row-major bf16 row for this lane's M-row.
// A-fragment layout (matches packed B): lanes 0-15 hold K = k0+[0..15],
// lanes 16-31 hold K = k0+16+[0..15]  -> kh = lane>>4 selects the +16 shift.
// B comes pre-swizzled: 16 contiguous bf16 per lane per fragment.
__device__ __forceinline__ v8f wmma_bf16(v8f c, const unsigned short* Arow, int k0, int kh,
                                         const unsigned short* Bfrag) {
    BF16Frag A, B;
    A.q[0] = *(const uint4*)(Arow + k0 + kh * 16);
    A.q[1] = *(const uint4*)(Arow + k0 + kh * 16 + 8);
    B.q[0] = *(const uint4*)(Bfrag);
    B.q[1] = *(const uint4*)(Bfrag + 8);
    return __builtin_amdgcn_wmma_f32_16x16x32_bf16(false, A.v, false, B.v, (short)0, c, false, false);
}

// ---------------- prep kernels ----------------

__global__ void k_convert_node(const float* __restrict__ nf, unsigned short* __restrict__ out) {
    int i = blockIdx.x * 256 + threadIdx.x;
    if (i < NNODE * HDIM) out[i] = f2bf(nf[i]);
}

// Pack (Krows x 128) fp32 weight into bf16 B-fragment layout:
// out[((ks*8+nt)*32 + lane)*16 + e] = W[ks*32 + (lane>>4)*16 + e][nt*16 + (lane&15)]
__global__ void k_pack_w(const float* __restrict__ W, unsigned short* __restrict__ out, int Krows) {
    int i = blockIdx.x * 256 + threadIdx.x;
    if (i >= Krows * HDIM) return;
    int e    = i & 15;
    int lane = (i >> 4) & 31;
    int frag = i >> 9;
    int nt   = frag & 7;
    int ks   = frag >> 3;
    int K    = ks * 32 + (lane >> 4) * 16 + e;
    int n    = nt * 16 + (lane & 15);
    out[i] = f2bf(W[K * HDIM + n]);
}

__global__ void k_init(unsigned* __restrict__ mmax, float* __restrict__ denom, float* __restrict__ acc) {
    int i = blockIdx.x * 256 + threadIdx.x;
    if (i < NNODE) { mmax[i] = 0u; denom[i] = 0.f; }
    if (i < NNODE * HDIM) acc[i] = 0.f;
}

// ---------------- h = node_feat @ W_node ; hd = h @ a_dst ----------------

__global__ __launch_bounds__(128) void k_h(const unsigned short* __restrict__ nodebf,
                                           const unsigned short* __restrict__ WnodeP,
                                           const float* __restrict__ a_dst,
                                           float* __restrict__ h, float* __restrict__ hd) {
    __shared__ float hout[16 * HDIM];
    __shared__ float red[128];
    int tid = threadIdx.x;
    int lane = tid & 31, wave = tid >> 5;
    int kh = lane >> 4;
    int n0 = blockIdx.x * 16;
    const unsigned short* Arow = nodebf + (size_t)(n0 + (lane & 15)) * HDIM;

    for (int t = 0; t < 2; ++t) {
        int nt = wave + 4 * t;
        v8f c = {0.f, 0.f, 0.f, 0.f, 0.f, 0.f, 0.f, 0.f};
        for (int ks = 0; ks < 4; ++ks)
            c = wmma_bf16(c, Arow, ks * 32, kh, WnodeP + ((ks * 8 + nt) * 32 + lane) * 16);
        int n = nt * 16 + (lane & 15);
        for (int r = 0; r < 8; ++r) {
            int m = kh * 8 + r;
            h[(size_t)(n0 + m) * HDIM + n] = c[r];
            hout[m * HDIM + n] = c[r];
        }
    }
    __syncthreads();
    int rr = tid >> 3, sub = tid & 7;
    float p = 0.f;
    for (int j = 0; j < 16; ++j) p += hout[rr * HDIM + sub * 16 + j] * a_dst[sub * 16 + j];
    red[tid] = p;
    __syncthreads();
    if (sub == 0) {
        float s = 0.f;
        for (int k = 0; k < 8; ++k) s += red[rr * 8 + k];
        hd[n0 + rr] = s;
    }
}

// ---------------- fused per-edge pipeline (32 edges / block, 8 waves) ----------------

__global__ __launch_bounds__(256) void k_edge(
    const unsigned short* __restrict__ nodebf, const float* __restrict__ edge_feat,
    const int* __restrict__ src, const int* __restrict__ dst,
    const float* __restrict__ dist, const float* __restrict__ h, const float* __restrict__ hd,
    const unsigned short* __restrict__ WaggP, const float* __restrict__ b_agg,
    const unsigned short* __restrict__ WegP, const unsigned short* __restrict__ WedgeP,
    const float* __restrict__ b_eg, const unsigned short* __restrict__ WeP,
    const float* __restrict__ a_src,
    float* __restrict__ out, unsigned short* __restrict__ msg_g, float* __restrict__ score_g,
    unsigned* __restrict__ mmax)
{
    __shared__ unsigned short X1[32 * 448];       // [h_src | h_dst | edge | dist] bf16
    __shared__ unsigned short enewb[32 * HDIM];
    __shared__ unsigned short egb[32 * HDIM];
    __shared__ float msgs[32 * HDIM];
    __shared__ float hsrc[32 * HDIM];
    __shared__ float red[256];
    __shared__ int sIdx[32], dIdx[32];
    __shared__ float hdl[32];

    int tid = threadIdx.x;
    int e0 = blockIdx.x * 32;
    if (tid < 32) { sIdx[tid] = src[e0 + tid]; dIdx[tid] = dst[e0 + tid]; }
    __syncthreads();
    if (tid < 32) hdl[tid] = hd[dIdx[tid]];

    // gather h_src / h_dst (bf16, 16B chunks) -- node data is L2-resident, keep cached
    for (int i = tid; i < 32 * 16; i += 256) {
        int row = i >> 4, q = i & 15;
        const uint4* sp = (const uint4*)(nodebf + (size_t)sIdx[row] * HDIM);
        const uint4* dp = (const uint4*)(nodebf + (size_t)dIdx[row] * HDIM);
        ((uint4*)&X1[row * 448])[q] = sp[q];
        ((uint4*)&X1[row * 448 + 128])[q] = dp[q];
    }
    // edge_feat -> bf16 (single-use stream: non-temporal)
    for (int i = tid; i < 32 * HDIM; i += 256) {
        int row = i >> 7, col = i & 127;
        X1[row * 448 + 256 + col] =
            f2bf(__builtin_nontemporal_load(&edge_feat[(size_t)(e0 + row) * HDIM + col]));
    }
    // dist -> bf16 (single-use stream: non-temporal)
    for (int i = tid; i < 32 * EDDIM; i += 256) {
        int row = i >> 6, col = i & 63;
        X1[row * 448 + 384 + col] =
            f2bf(__builtin_nontemporal_load(&dist[(size_t)(e0 + row) * EDDIM + col]));
    }
    // gather h[src] fp32 (L2-resident)
    for (int i = tid; i < 32 * 32; i += 256) {
        int row = i >> 5, q = i & 31;
        ((float4*)&hsrc[row * HDIM])[q] = ((const float4*)(h + (size_t)sIdx[row] * HDIM))[q];
    }
    __syncthreads();

    int lane = tid & 31, wave = tid >> 5;      // 8 waves
    int kh = lane >> 4;
    int mt = wave & 1;                          // M tile: rows mt*16 .. mt*16+15
    int ntb = wave >> 1;                        // N tiles: ntb, ntb+4
    int rbase = mt * 16 + (lane & 15);
    const unsigned short* Arow  = &X1[rbase * 448];
    const unsigned short* ArowE = &enewb[rbase * HDIM];
    const unsigned short* ArowG = &egb[rbase * HDIM];

    // GEMM1: e_new = relu(X1 @ W_agg + b_agg), K = 448
    for (int t = 0; t < 2; ++t) {
        int nt = ntb + 4 * t;
        v8f c = {0.f, 0.f, 0.f, 0.f, 0.f, 0.f, 0.f, 0.f};
        for (int ks = 0; ks < 14; ++ks)
            c = wmma_bf16(c, Arow, ks * 32, kh, WaggP + ((ks * 8 + nt) * 32 + lane) * 16);
        int n = nt * 16 + (lane & 15);
        float bias = b_agg[n];
        for (int r = 0; r < 8; ++r) {
            int m = mt * 16 + kh * 8 + r;
            float v = c[r] + bias;
            v = v > 0.f ? v : 0.f;
            __builtin_nontemporal_store(v, &out[(size_t)(NNODE + e0 + m) * HDIM + n]);
            enewb[m * HDIM + n] = f2bf(v);
        }
    }
    __syncthreads();

    // GEMM2: eg = relu([h_src|h_dst] @ W_eg + e_new @ W_edge_eg + b_eg)
    for (int t = 0; t < 2; ++t) {
        int nt = ntb + 4 * t;
        v8f c = {0.f, 0.f, 0.f, 0.f, 0.f, 0.f, 0.f, 0.f};
        for (int ks = 0; ks < 8; ++ks)
            c = wmma_bf16(c, Arow, ks * 32, kh, WegP + ((ks * 8 + nt) * 32 + lane) * 16);
        for (int ks = 0; ks < 4; ++ks)
            c = wmma_bf16(c, ArowE, ks * 32, kh, WedgeP + ((ks * 8 + nt) * 32 + lane) * 16);
        int n = nt * 16 + (lane & 15);
        float bias = b_eg[n];
        for (int r = 0; r < 8; ++r) {
            int m = mt * 16 + kh * 8 + r;
            float v = c[r] + bias;
            v = v > 0.f ? v : 0.f;
            egb[m * HDIM + n] = f2bf(v);
        }
    }
    __syncthreads();

    // GEMM3: msg = h[src] + eg @ W_e  (spill msg as bf16, non-temporal)
    for (int t = 0; t < 2; ++t) {
        int nt = ntb + 4 * t;
        v8f c = {0.f, 0.f, 0.f, 0.f, 0.f, 0.f, 0.f, 0.f};
        for (int ks = 0; ks < 4; ++ks)
            c = wmma_bf16(c, ArowG, ks * 32, kh, WeP + ((ks * 8 + nt) * 32 + lane) * 16);
        int n = nt * 16 + (lane & 15);
        for (int r = 0; r < 8; ++r) {
            int m = mt * 16 + kh * 8 + r;
            float v = c[r] + hsrc[m * HDIM + n];
            msgs[m * HDIM + n] = v;
            __builtin_nontemporal_store(f2bf(v), &msg_g[(size_t)(e0 + m) * HDIM + n]);
        }
    }
    __syncthreads();

    // score = leaky_relu(msg . a_src + hd[dst]); segment max via ordered-uint atomicMax
    int rr = tid >> 3, sub = tid & 7;           // 32 rows x 8 partials
    float p = 0.f;
    for (int j = 0; j < 16; ++j) p += msgs[rr * HDIM + sub * 16 + j] * a_src[sub * 16 + j];
    red[tid] = p;
    __syncthreads();
    if (sub == 0) {
        float s = hdl[rr];
        for (int k = 0; k < 8; ++k) s += red[rr * 8 + k];
        s = s > 0.f ? s : 0.2f * s;
        __builtin_nontemporal_store(s, &score_g[e0 + rr]);
        unsigned u = __float_as_uint(s);
        u = (u & 0x80000000u) ? ~u : (u | 0x80000000u);
        atomicMax(&mmax[dIdx[rr]], u);
    }
}

// ---------------- softmax accumulate + finalize ----------------

__global__ __launch_bounds__(256) void k_pass2(const int* __restrict__ dst,
                                               const float* __restrict__ score,
                                               const unsigned short* __restrict__ msg_g,
                                               const unsigned* __restrict__ mmax,
                                               float* __restrict__ denom, float* __restrict__ acc) {
    int tid = threadIdx.x;
    int e = blockIdx.x * 2 + (tid >> 7);
    int col = tid & 127;
    int d = dst[e];
    unsigned u = mmax[d];
    float mx = (u & 0x80000000u) ? __uint_as_float(u & 0x7FFFFFFFu) : __uint_as_float(~u);
    float w = __expf(score[e] - mx);
    if (col == 0) atomicAdd(&denom[d], w);
    float mv = bf2f(__builtin_nontemporal_load(&msg_g[(size_t)e * HDIM + col]));
    atomicAdd(&acc[(size_t)d * HDIM + col], w * mv);
}

__global__ void k_final(const float* __restrict__ acc, const float* __restrict__ denom,
                        const float* __restrict__ b_node, float* __restrict__ out) {
    int i = blockIdx.x * 256 + threadIdx.x;
    if (i >= NNODE * HDIM) return;
    float v = acc[i] / (denom[i >> 7] + 1e-16f) + b_node[i & 127];
    v = v > 0.f ? v : 0.f;
    __builtin_nontemporal_store(v, &out[i]);
}

// ---------------- launcher ----------------

extern "C" void kernel_launch(void* const* d_in, const int* in_sizes, int n_in,
                              void* d_out, int out_size, void* d_ws, size_t ws_size,
                              hipStream_t stream) {
    const float* node_feat = (const float*)d_in[0];
    const float* edge_feat = (const float*)d_in[1];
    const int*   edge_idx  = (const int*)d_in[2];
    const float* dist_feat = (const float*)d_in[3];
    const float* W_agg  = (const float*)d_in[4];
    const float* b_agg  = (const float*)d_in[5];
    const float* W_eg   = (const float*)d_in[6];
    const float* W_edge = (const float*)d_in[7];
    const float* b_eg   = (const float*)d_in[8];
    const float* W_node = (const float*)d_in[9];
    const float* W_e    = (const float*)d_in[10];
    const float* a_src  = (const float*)d_in[11];
    const float* a_dst  = (const float*)d_in[12];
    const float* b_node = (const float*)d_in[13];
    float* out = (float*)d_out;
    const int* src = edge_idx;
    const int* dst = edge_idx + NEDGE;

    char* base = (char*)d_ws;
    size_t o = 0;
    auto alloc = [&](size_t bytes) { size_t r = o; o = (o + bytes + 255) & ~(size_t)255; return r; };
    size_t off_nodebf = alloc((size_t)NNODE * HDIM * 2);
    size_t off_h      = alloc((size_t)NNODE * HDIM * 4);
    size_t off_hd     = alloc((size_t)NNODE * 4);
    size_t off_mmax   = alloc((size_t)NNODE * 4);
    size_t off_denom  = alloc((size_t)NNODE * 4);
    size_t off_acc    = alloc((size_t)NNODE * HDIM * 4);
    size_t off_msg    = alloc((size_t)NEDGE * HDIM * 2);   // bf16 spill
    size_t off_score  = alloc((size_t)NEDGE * 4);
    size_t off_wagg   = alloc((size_t)448 * HDIM * 2);
    size_t off_weg    = alloc((size_t)256 * HDIM * 2);
    size_t off_wedge  = alloc((size_t)HDIM * HDIM * 2);
    size_t off_wnode  = alloc((size_t)HDIM * HDIM * 2);
    size_t off_we     = alloc((size_t)HDIM * HDIM * 2);
    (void)ws_size;

    unsigned short* nodebf = (unsigned short*)(base + off_nodebf);
    float*    hbuf  = (float*)(base + off_h);
    float*    hd    = (float*)(base + off_hd);
    unsigned* mmax  = (unsigned*)(base + off_mmax);
    float*    denom = (float*)(base + off_denom);
    float*    acc   = (float*)(base + off_acc);
    unsigned short* msg_g = (unsigned short*)(base + off_msg);
    float*    score = (float*)(base + off_score);
    unsigned short* WaggP  = (unsigned short*)(base + off_wagg);
    unsigned short* WegP   = (unsigned short*)(base + off_weg);
    unsigned short* WedgeP = (unsigned short*)(base + off_wedge);
    unsigned short* WnodeP = (unsigned short*)(base + off_wnode);
    unsigned short* WeP    = (unsigned short*)(base + off_we);

    const int NH = NNODE * HDIM;
    k_convert_node<<<(NH + 255) / 256, 256, 0, stream>>>(node_feat, nodebf);
    k_pack_w<<<(448 * HDIM + 255) / 256, 256, 0, stream>>>(W_agg,  WaggP,  448);
    k_pack_w<<<(256 * HDIM + 255) / 256, 256, 0, stream>>>(W_eg,   WegP,   256);
    k_pack_w<<<(HDIM * HDIM + 255) / 256, 256, 0, stream>>>(W_edge, WedgeP, HDIM);
    k_pack_w<<<(HDIM * HDIM + 255) / 256, 256, 0, stream>>>(W_node, WnodeP, HDIM);
    k_pack_w<<<(HDIM * HDIM + 255) / 256, 256, 0, stream>>>(W_e,    WeP,    HDIM);
    k_init<<<(NH + 255) / 256, 256, 0, stream>>>(mmax, denom, acc);

    k_h<<<NNODE / 16, 128, 0, stream>>>(nodebf, WnodeP, a_dst, hbuf, hd);

    k_edge<<<NEDGE / 32, 256, 0, stream>>>(nodebf, edge_feat, src, dst, dist_feat, hbuf, hd,
                                           WaggP, b_agg, WegP, WedgeP, b_eg, WeP, a_src,
                                           out, msg_g, score, mmax);

    k_pass2<<<NEDGE / 2, 256, 0, stream>>>(dst, score, msg_g, mmax, denom, acc);
    k_final<<<(NH + 255) / 256, 256, 0, stream>>>(acc, denom, b_node, out);
}
